// MH_attention_50010599195063
// MI455X (gfx1250) — matrix operand order repaired and checked by reading, not compiled, not run on hardware
//
#include <hip/hip_runtime.h>
#include <hip/hip_bf16.h>
#include <stdint.h>

// ---------------------------------------------------------------------------
// MI455X (gfx1250) multi-head attention, bf16 WMMA.
//   x[4,2048,768] @ w_qkv[768,2304] + b -> per-head Q/K/V bf16 (Q pre-scaled
//   by log2e/sqrt(64) so softmax is pure exp2)
//   flash attention: 128 q-rows/WG (8 waves), 64-key blocks, double-buffered
//   attn[8192,768] @ w_proj + b -> f32 d_out
// ~90 GFLOP vs ~100MB traffic -> matrix-op bound. Every WMMA fragment is two
// ds_load_b128; softmax reductions are DPP (VALU), not ds_bpermute.
// ---------------------------------------------------------------------------

typedef __bf16 bf16;
typedef __bf16 v16bf __attribute__((ext_vector_type(16)));
typedef __bf16 v8bf  __attribute__((ext_vector_type(8)));
typedef __bf16 v2bf  __attribute__((ext_vector_type(2)));
typedef __bf16 v4bf  __attribute__((ext_vector_type(4)));
typedef float  v8f   __attribute__((ext_vector_type(8)));

__device__ __forceinline__ v8f wmma_bf16(v16bf a, v16bf b, v8f c) {
    return __builtin_amdgcn_wmma_f32_16x16x32_bf16(
        false, a, false, b, (short)0, c, false, false);
}

__device__ __forceinline__ v16bf join8(v8bf lo, v8bf hi) {
    return __builtin_shufflevector(lo, hi,
        0, 1, 2, 3, 4, 5, 6, 7, 8, 9, 10, 11, 12, 13, 14, 15);
}
// A-frag (16x32): per lane 8 bf16 at p, 8 at p+16 (p includes (lane&16)?8:0)
__device__ __forceinline__ v16bf frag_split16(const bf16* p) {
    return join8(*(const v8bf*)p, *(const v8bf*)(p + 16));
}
// B-frag (32x16) from [N][K] tile: 8 bf16 at p, 8 at p+16 (p includes (lane&16)?8:0)
__device__ __forceinline__ v16bf frag_contig16(const bf16* p) {
    return join8(*(const v8bf*)p, *(const v8bf*)(p + 16));
}
// pack two bf16 (adjacent K) into one ds_store_b32
__device__ __forceinline__ void store_pair(bf16* dst, bf16 a, bf16 b) {
    v2bf v; v[0] = a; v[1] = b;
    *(v2bf*)dst = v;
}

// ---- DPP reductions across each 16-lane group (VALU, no LDS traffic) -------
template <int CTRL>
__device__ __forceinline__ float dpp_mov(float x) {
    return __int_as_float(__builtin_amdgcn_update_dpp(
        0, __float_as_int(x), CTRL, 0xF, 0xF, true));
}
__device__ __forceinline__ float red_max16(float x) {
    x = fmaxf(x, dpp_mov<0xB1>(x));    // quad_perm(1,0,3,2)  xor1
    x = fmaxf(x, dpp_mov<0x4E>(x));    // quad_perm(2,3,0,1)  xor2
    x = fmaxf(x, dpp_mov<0x141>(x));   // row_half_mirror     (8-group)
    x = fmaxf(x, dpp_mov<0x140>(x));   // row_mirror          (16-group)
    return x;
}
__device__ __forceinline__ float red_add16(float x) {
    x += dpp_mov<0xB1>(x);
    x += dpp_mov<0x4E>(x);
    x += dpp_mov<0x141>(x);
    x += dpp_mov<0x140>(x);
    return x;
}

// ---------------------------------------------------------------------------
__global__ void cvt_f32_bf16_kernel(const float* __restrict__ in,
                                    bf16* __restrict__ out, int n4) {
    int i = blockIdx.x * blockDim.x + threadIdx.x;
    if (i < n4) {
        float4 f = ((const float4*)in)[i];
        v4bf o;
        o[0] = (bf16)f.x; o[1] = (bf16)f.y; o[2] = (bf16)f.z; o[3] = (bf16)f.w;
        ((v4bf*)out)[i] = o;
    }
}

// ---------------------------------------------------------------------------
// GEMM: C[M,N] = A[M,K] * B[K,N] + bias. 256 thr = 8 waves.
// BM=128 BN=128 BK=32, double-buffered LDS; wave w: rows (w&3)*32, cols (w>>2)*64.
#define MODE_QKV  0
#define MODE_PROJ 1

template <int MODE>
__global__ __launch_bounds__(256) void gemm_bf16_kernel(
    const bf16* __restrict__ A, const bf16* __restrict__ B,
    const float* __restrict__ bias,
    float* __restrict__ outF,
    bf16* __restrict__ outQ, bf16* __restrict__ outK, bf16* __restrict__ outV,
    int M, int N, int K) {
    __shared__ bf16 As[2][128 * 32];   // [m][k]
    __shared__ bf16 BsT[2][128 * 32];  // [n][k] -> contiguous-K B-frags

    const int tid  = threadIdx.x;
    const int lane = tid & 31;
    const int w    = tid >> 5;
    const int m0   = blockIdx.y * 128;
    const int n0   = blockIdx.x * 128;
    const int mr   = (w & 3) * 32;
    const int nc   = (w >> 2) * 64;

    const int ar  = tid >> 1, ac = (tid & 1) * 16;          // A: 16 elems/thr
    const int bkr = (tid & 15) * 2, bcg = (tid >> 4) * 8;   // B: 2 rows x 8 cols

    const int nk = K / 32;
    v8f acc[2][4] = {};

    // prologue: stage tile 0 into buffer 0
    {
        const bf16* asrc = &A[(size_t)(m0 + ar) * K + ac];
        *(uint4*)&As[0][ar * 32 + ac]     = *(const uint4*)&asrc[0];
        *(uint4*)&As[0][ar * 32 + ac + 8] = *(const uint4*)&asrc[8];
        v8bf r0 = *(const v8bf*)&B[(size_t)(bkr)     * N + n0 + bcg];
        v8bf r1 = *(const v8bf*)&B[(size_t)(bkr + 1) * N + n0 + bcg];
#pragma unroll
        for (int j = 0; j < 8; ++j)
            store_pair(&BsT[0][(bcg + j) * 32 + bkr], r0[j], r1[j]);
    }
    __syncthreads();

    const int akhi = (lane & 16) ? 8 : 0;
    const int bkhi = (lane & 16) ? 8 : 0;

    for (int ks = 0; ks < nk; ++ks) {
        const int buf = ks & 1;
        const bool has_next = (ks + 1 < nk);
        uint4 a0reg, a1reg; v8bf b0reg, b1reg;
        if (ks + 2 < nk) {   // distance-2 prefetch (global_prefetch_b8)
            __builtin_prefetch(&A[(size_t)(m0 + ar) * K + (ks + 2) * 32 + ac], 0, 1);
            __builtin_prefetch(&B[(size_t)((ks + 2) * 32 + bkr) * N + n0 + bcg], 0, 1);
        }
        if (has_next) {      // issue next tile's global loads before compute
            const int k1 = (ks + 1) * 32;
            const bf16* asrc = &A[(size_t)(m0 + ar) * K + k1 + ac];
            a0reg = *(const uint4*)&asrc[0];
            a1reg = *(const uint4*)&asrc[8];
            b0reg = *(const v8bf*)&B[(size_t)(k1 + bkr)     * N + n0 + bcg];
            b1reg = *(const v8bf*)&B[(size_t)(k1 + bkr + 1) * N + n0 + bcg];
        }

        v16bf af0 = frag_split16(&As[buf][(mr + (lane & 15)) * 32 + akhi]);
        v16bf af1 = frag_split16(&As[buf][(mr + 16 + (lane & 15)) * 32 + akhi]);
#pragma unroll
        for (int nt = 0; nt < 4; ++nt) {
            v16bf bf = frag_contig16(&BsT[buf][(nc + nt * 16 + (lane & 15)) * 32 + bkhi]);
            acc[0][nt] = wmma_bf16(af0, bf, acc[0][nt]);
            acc[1][nt] = wmma_bf16(af1, bf, acc[1][nt]);
        }

        if (has_next) {
            const int nb = buf ^ 1;
            *(uint4*)&As[nb][ar * 32 + ac]     = a0reg;
            *(uint4*)&As[nb][ar * 32 + ac + 8] = a1reg;
#pragma unroll
            for (int j = 0; j < 8; ++j)
                store_pair(&BsT[nb][(bcg + j) * 32 + bkr], b0reg[j], b1reg[j]);
        }
        __syncthreads();
    }

    // epilogue: C-frag -> lane holds col lane&15, rows c + 8*(lane>=16)
    const float qscale = 1.44269504f * 0.125f;   // log2e / sqrt(64)
    const int n_l = lane & 15;
    const int hi  = (lane & 16) ? 8 : 0;
#pragma unroll
    for (int mt = 0; mt < 2; ++mt) {
#pragma unroll
        for (int nt = 0; nt < 4; ++nt) {
            int col = n0 + nc + nt * 16 + n_l;
            float bv = bias[col];
#pragma unroll
            for (int c = 0; c < 8; ++c) {
                int   row = m0 + mr + mt * 16 + c + hi;
                float v   = acc[mt][nt][c] + bv;
                if (MODE == MODE_PROJ) {
                    outF[(size_t)row * N + col] = v;
                } else {
                    int which = col / 768;
                    int rem   = col - which * 768;
                    int h     = rem >> 6;
                    int d     = rem & 63;
                    int bb    = row >> 11;
                    int sl    = row & 2047;
                    if (which == 0) v *= qscale;   // fold softmax scale into Q
                    bf16* dst = (which == 0) ? outQ : (which == 1) ? outK : outV;
                    dst[(((size_t)(bb * 12 + h)) * 2048 + sl) * 64 + d] = (bf16)v;
                }
            }
        }
    }
}

// ---------------------------------------------------------------------------
// Flash attention: grid = (B*H)*(S/128), 256 thr = 8 waves, 16 q-rows/wave,
// 64-key blocks, double-buffered K/V staging.
__global__ __launch_bounds__(256) void attn_flash_kernel(
    const bf16* __restrict__ Q, const bf16* __restrict__ Kg,
    const bf16* __restrict__ V, bf16* __restrict__ Out) {
    constexpr int S = 2048, HD = 64, KB = 64, NKB = S / KB;
    __shared__ bf16 Qs[128 * 64];       // [qrow][d]
    __shared__ bf16 Ks[2][KB * 64];     // [key][d]  (K^T frags: contiguous d)
    __shared__ bf16 VsT[2][64 * KB];    // [d][key]  (V frags: contiguous key)
    __shared__ bf16 Pbf[8][16 * KB];    // per-wave P, A-frag row-major

    const int bh   = blockIdx.x >> 4;
    const int q0   = (blockIdx.x & 15) * 128;
    const int tid  = threadIdx.x;
    const int lane = tid & 31;
    const int w    = tid >> 5;

    const bf16* Qb = Q  + (size_t)bh * S * HD;
    const bf16* Kb = Kg + (size_t)bh * S * HD;
    const bf16* Vb = V  + (size_t)bh * S * HD;

    // stage Q block [128][64]
    {
        int r = tid >> 1, cb = (tid & 1) * 32;
#pragma unroll
        for (int c = 0; c < 4; ++c)
            *(uint4*)&Qs[r * 64 + cb + c * 8] =
                *(const uint4*)&Qb[(size_t)(q0 + r) * 64 + cb + c * 8];
    }
    __syncthreads();

    const int akhi = (lane & 16) ? 8 : 0;
    const int bkhi = (lane & 16) ? 8 : 0;
    const bf16* qrow = &Qs[(w * 16 + (lane & 15)) * 64 + akhi];
    v16bf aq0 = frag_split16(qrow);        // Q pre-scaled by log2e/8
    v16bf aq1 = frag_split16(qrow + 32);

    float mrow[8], lrow[8];
#pragma unroll
    for (int i = 0; i < 8; ++i) { mrow[i] = -1e30f; lrow[i] = 0.f; }
    v8f o0 = {}, o1 = {}, o2 = {}, o3 = {};

    // staging coords: K 16 elems/thr; V 2 keys x 8 dims/thr
    const int krow = tid >> 2, kcol = (tid & 3) * 16;
    const int vk = (tid & 31) * 2, vc = (tid >> 5) * 8;

    // prologue: stage block 0 into buffer 0
    {
        const bf16* ksrc = &Kb[(size_t)krow * 64 + kcol];
        *(uint4*)&Ks[0][krow * 64 + kcol]     = *(const uint4*)&ksrc[0];
        *(uint4*)&Ks[0][krow * 64 + kcol + 8] = *(const uint4*)&ksrc[8];
        const bf16* vsrc = &Vb[(size_t)vk * 64 + vc];
        v8bf r0 = *(const v8bf*)vsrc;
        v8bf r1 = *(const v8bf*)(vsrc + 64);
#pragma unroll
        for (int j = 0; j < 8; ++j)
            store_pair(&VsT[0][(vc + j) * KB + vk], r0[j], r1[j]);
    }
    __syncthreads();

    for (int kb = 0; kb < NKB; ++kb) {
        const int  buf = kb & 1;
        const bool has_next = (kb + 1 < NKB);
        uint4 k0reg, k1reg; v8bf v0reg, v1reg;
        if (kb + 2 < NKB) {
            __builtin_prefetch(&Kb[(size_t)((kb + 2) * KB + krow) * 64 + kcol], 0, 1);
            __builtin_prefetch(&Vb[(size_t)((kb + 2) * KB + vk) * 64 + vc], 0, 1);
        }
        if (has_next) {
            const bf16* ksrc = &Kb[(size_t)((kb + 1) * KB + krow) * 64 + kcol];
            k0reg = *(const uint4*)&ksrc[0];
            k1reg = *(const uint4*)&ksrc[8];
            const bf16* vsrc = &Vb[(size_t)((kb + 1) * KB + vk) * 64 + vc];
            v0reg = *(const v8bf*)vsrc;
            v1reg = *(const v8bf*)(vsrc + 64);
        }
        const bf16* KsB = Ks[buf];
        const bf16* VtB = VsT[buf];

        // scores S[16q x 64k]: 4 key groups x 2 head-dim chunks
        v8f s[4] = {};
#pragma unroll
        for (int g = 0; g < 4; ++g) {
            const bf16* col = &KsB[(g * 16 + (lane & 15)) * 64 + bkhi];
            s[g] = wmma_bf16(aq0, frag_contig16(col), s[g]);
            s[g] = wmma_bf16(aq1, frag_contig16(col + 32), s[g]);
        }

        // online softmax (DPP reductions across the 16-lane half)
        float alpha[8];
#pragma unroll
        for (int c = 0; c < 8; ++c) {
            float mx = fmaxf(fmaxf(s[0][c], s[1][c]), fmaxf(s[2][c], s[3][c]));
            mx = red_max16(mx);
            float mnew = fmaxf(mrow[c], mx);
            alpha[c]   = exp2f(mrow[c] - mnew);
            mrow[c]    = mnew;
            float p0 = exp2f(s[0][c] - mnew);
            float p1 = exp2f(s[1][c] - mnew);
            float p2 = exp2f(s[2][c] - mnew);
            float p3 = exp2f(s[3][c] - mnew);
            s[0][c] = p0; s[1][c] = p1; s[2][c] = p2; s[3][c] = p3;
            float rs = red_add16((p0 + p1) + (p2 + p3));
            lrow[c] = lrow[c] * alpha[c] + rs;
            o0[c] *= alpha[c]; o1[c] *= alpha[c];
            o2[c] *= alpha[c]; o3[c] *= alpha[c];
        }

        // P -> per-wave LDS (bf16, A-frag row-major [16][64])
        bf16* Pw = Pbf[w];
        {
            int n = lane & 15, hi8 = (lane & 16) ? 8 : 0;
#pragma unroll
            for (int c = 0; c < 8; ++c) {
#pragma unroll
                for (int g = 0; g < 4; ++g)
                    Pw[(c + hi8) * KB + g * 16 + n] = (bf16)s[g][c];
            }
        }
        v16bf ap0 = frag_split16(&Pw[(lane & 15) * KB + akhi]);
        v16bf ap1 = frag_split16(&Pw[(lane & 15) * KB + 32 + akhi]);

        // O += P(16x64) . V(64x64): 4 dim groups x 2 key chunks
        {
            const bf16* c0 = &VtB[((lane & 15)) * KB + bkhi];
            o0 = wmma_bf16(ap0, frag_contig16(c0), o0);
            o0 = wmma_bf16(ap1, frag_contig16(c0 + 32), o0);
            const bf16* c1 = &VtB[(16 + (lane & 15)) * KB + bkhi];
            o1 = wmma_bf16(ap0, frag_contig16(c1), o1);
            o1 = wmma_bf16(ap1, frag_contig16(c1 + 32), o1);
            const bf16* c2 = &VtB[(32 + (lane & 15)) * KB + bkhi];
            o2 = wmma_bf16(ap0, frag_contig16(c2), o2);
            o2 = wmma_bf16(ap1, frag_contig16(c2 + 32), o2);
            const bf16* c3 = &VtB[(48 + (lane & 15)) * KB + bkhi];
            o3 = wmma_bf16(ap0, frag_contig16(c3), o3);
            o3 = wmma_bf16(ap1, frag_contig16(c3 + 32), o3);
        }

        if (has_next) {   // stage next block into the other buffer
            const int nb = buf ^ 1;
            *(uint4*)&Ks[nb][krow * 64 + kcol]     = k0reg;
            *(uint4*)&Ks[nb][krow * 64 + kcol + 8] = k1reg;
#pragma unroll
            for (int j = 0; j < 8; ++j)
                store_pair(&VsT[nb][(vc + j) * KB + vk], v0reg[j], v1reg[j]);
        }
        __syncthreads();
    }

    // normalize + scatter to [B,S,768] (fuses head transpose for proj GEMM)
    const int b = bh / 12, h = bh % 12;
    const int n = lane & 15, hi8 = (lane & 16) ? 8 : 0;
#pragma unroll
    for (int c = 0; c < 8; ++c) {
        float inv = (lrow[c] > 0.f) ? 1.f / lrow[c] : 0.f;
        int    sr = q0 + w * 16 + c + hi8;
        size_t base = ((size_t)b * S + sr) * 768 + h * 64;
        Out[base + n]      = (bf16)(o0[c] * inv);
        Out[base + 16 + n] = (bf16)(o1[c] * inv);
        Out[base + 32 + n] = (bf16)(o2[c] * inv);
        Out[base + 48 + n] = (bf16)(o3[c] * inv);
    }
}

// ---------------------------------------------------------------------------
extern "C" void kernel_launch(void* const* d_in, const int* in_sizes, int n_in,
                              void* d_out, int out_size, void* d_ws, size_t ws_size,
                              hipStream_t stream) {
    (void)in_sizes; (void)n_in; (void)out_size; (void)ws_size;
    constexpr int B = 4, S = 2048, D = 768, H = 12;
    constexpr int M = B * S;   // 8192

    const float* x      = (const float*)d_in[0];
    const float* w_qkv  = (const float*)d_in[1];
    const float* b_qkv  = (const float*)d_in[2];
    const float* w_proj = (const float*)d_in[3];
    const float* b_proj = (const float*)d_in[4];
    float* out = (float*)d_out;

    uint8_t* ws = (uint8_t*)d_ws;
    size_t off = 0;
    auto carve = [&](size_t elems) {
        bf16* p = (bf16*)(ws + off);
        off += (elems * sizeof(bf16) + 255) & ~(size_t)255;
        return p;
    };
    bf16* xbf  = carve((size_t)M * D);
    bf16* wqbf = carve((size_t)D * 3 * D);
    bf16* wpbf = carve((size_t)D * D);
    bf16* qbuf = carve((size_t)B * H * S * 64);
    bf16* kbuf = carve((size_t)B * H * S * 64);
    bf16* vbuf = carve((size_t)B * H * S * 64);
    bf16* attn = carve((size_t)M * D);

    auto cvt = [&](const float* src, bf16* dst, int n) {
        cvt_f32_bf16_kernel<<<(n / 4 + 255) / 256, 256, 0, stream>>>(src, dst, n / 4);
    };
    cvt(x,      xbf,  M * D);
    cvt(w_qkv,  wqbf, D * 3 * D);
    cvt(w_proj, wpbf, D * D);

    gemm_bf16_kernel<MODE_QKV>
        <<<dim3(3 * D / 128, M / 128), 256, 0, stream>>>(
            xbf, wqbf, b_qkv, nullptr, qbuf, kbuf, vbuf, M, 3 * D, D);

    attn_flash_kernel<<<B * H * (S / 128), 256, 0, stream>>>(
        qbuf, kbuf, vbuf, attn);

    gemm_bf16_kernel<MODE_PROJ>
        <<<dim3(D / 128, M / 128), 256, 0, stream>>>(
            attn, wpbf, b_proj, out, nullptr, nullptr, nullptr, M, D, D);
}
